// ContextAwareMissingEmbeddingGenerator_29764123361488
// MI455X (gfx1250) — matrix operand order repaired and checked by reading, not compiled, not run on hardware
//
#include <hip/hip_runtime.h>

// ---------------------------------------------------------------------------
// ContextAwareMissingEmbeddingGenerator for MI455X (gfx1250, wave32, WMMA)
//   B=4096, S=23, D=768, H=8, HD=96, L=50
// Big GEMMs via v_wmma_f32_16x16x32_bf16 with double-buffered async
// global->LDS staging (ASYNCcnt); tiny S=23 attention in f32 VALU.
// ---------------------------------------------------------------------------

typedef __attribute__((ext_vector_type(16))) __bf16 v16bf;
typedef __attribute__((ext_vector_type(8)))  float  v8f;
typedef int v4i_gcc __attribute__((__vector_size__(16)));   // matches builtin proto

constexpr int B_  = 4096;
constexpr int S_  = 23;
constexpr int D_  = 768;
constexpr int H_  = 8;
constexpr int HD_ = 96;
constexpr int L_  = 50;
constexpr int TD_ = 3 * D_;   // 2304

// ---- CDNA5 async global->LDS (bypasses VGPRs, tracked with ASYNCcnt) ------
// Probe-confirmed signature: (v4i __device__ *src, v4i __shared-AS *dst, Ii, Ii)
#if __has_builtin(__builtin_amdgcn_global_load_async_to_lds_b128)
#define HAVE_ASYNC_LDS 1
__device__ __forceinline__ void async_cp16(void* lds_dst, const void* gsrc) {
  __builtin_amdgcn_global_load_async_to_lds_b128(
      (__attribute__((address_space(1))) v4i_gcc*)gsrc,
      (__attribute__((address_space(3))) v4i_gcc*)lds_dst,
      0, 0);
}
#else
#define HAVE_ASYNC_LDS 0
#endif

__device__ __forceinline__ void wait_async_le4() {
#if __has_builtin(__builtin_amdgcn_s_wait_asynccnt)
  __builtin_amdgcn_s_wait_asynccnt(4);
#elif HAVE_ASYNC_LDS
  asm volatile("s_wait_asynccnt 0x4" ::: "memory");
#endif
}
__device__ __forceinline__ void wait_async_le0() {
#if __has_builtin(__builtin_amdgcn_s_wait_asynccnt)
  __builtin_amdgcn_s_wait_asynccnt(0);
#elif HAVE_ASYNC_LDS
  asm volatile("s_wait_asynccnt 0x0" ::: "memory");
#endif
}

// ---------------------------------------------------------------- utilities
__global__ void k_f32_to_bf16(const float* __restrict__ src,
                              __bf16* __restrict__ dst, int n) {
  int i = blockIdx.x * blockDim.x + threadIdx.x;
  if (i < n) dst[i] = (__bf16)src[i];
}

// sec = mask ? cls : missing_table  (stored bf16 for WMMA GEMM consumption)
__global__ void k_build_sec(const float* __restrict__ cls,
                            const float* __restrict__ miss,
                            const int*   __restrict__ mask,
                            __bf16* __restrict__ secbf, int total) {
  int i = blockIdx.x * blockDim.x + threadIdx.x;
  if (i >= total) return;
  int d  = i % D_;
  int bs = i / D_;
  int s  = bs % S_;
  float v = mask[bs] ? cls[i] : miss[s * D_ + d];
  secbf[i] = (__bf16)v;
}

// ------------------------------------------------------------- WMMA GEMM
// C[M,N](f32 and/or bf16) = A[M,K](bf16, row-major) @ W[N,K](bf16, row-major)^T + bias
// Block tile 128x128, K-step 32, 8 waves; wave tile 32x64 (2x4 wmma accums).
// Double-buffered LDS; tiles staged with async global->LDS when available.
// M % 128 == 0 and K % 32 == 0 guaranteed by caller; N may be ragged
// (handled by clamping B row index -> EXEC stays all-ones, store guarded).
#define BM 128
#define BN 128
#define BK 32

__launch_bounds__(256)
__global__ void k_gemm_bf16(const __bf16* __restrict__ A,
                            const __bf16* __restrict__ W,
                            const float*  __restrict__ bias,
                            float*  __restrict__ Cf,
                            __bf16* __restrict__ Cbf,
                            int M, int N, int K) {
  __shared__ __align__(16) __bf16 As[2][BM][BK];
  __shared__ __align__(16) __bf16 Bs[2][BN][BK];

  const int tid  = threadIdx.x;
  const int lane = tid & 31;
  const int wave = tid >> 5;      // 0..7
  const int wm   = wave >> 1;     // 0..3 : 32-row slab
  const int wn   = wave & 1;      // 0..1 : 64-col slab
  const int lrow = lane & 15;
  const int lhi  = lane >> 4;     // 0 or 1

  const int tileM = blockIdx.y * BM;
  const int tileN = blockIdx.x * BN;

  v8f zacc = {};
  v8f acc[2][4];
#pragma unroll
  for (int i = 0; i < 2; ++i)
#pragma unroll
    for (int j = 0; j < 4; ++j) acc[i][j] = zacc;

  union Frag { v16bf v; uint4 u[2]; };

  // stage one 128x32 A tile + 128x32 B tile into LDS buffer `buf`
  // (4 instructions per thread -> per-wave ASYNCcnt += 4 in async mode)
  auto issue_tile = [&](int buf, int k0) {
#pragma unroll
    for (int i = 0; i < 2; ++i) {
      int lin = tid + i * 256;                 // 0..511
      int r   = lin >> 2;                      // 0..127
      int c8  = (lin & 3) * 8;                 // bf16 column {0,8,16,24}
      const __bf16* ga = &A[(size_t)(tileM + r) * K + k0 + c8];
      int gn = tileN + r;
      if (gn > N - 1) gn = N - 1;              // clamp: keep EXEC all-ones
      const __bf16* gb = &W[(size_t)gn * K + k0 + c8];
#if HAVE_ASYNC_LDS
      async_cp16(&As[buf][r][c8], ga);
      async_cp16(&Bs[buf][r][c8], gb);
#else
      *(uint4*)&As[buf][r][c8] = *(const uint4*)ga;
      *(uint4*)&Bs[buf][r][c8] = *(const uint4*)gb;
#endif
    }
  };

  const int NT = K / BK;
  issue_tile(0, 0);

  for (int kt = 0; kt < NT; ++kt) {
    const int cur = kt & 1;
    if (kt + 1 < NT) {
      issue_tile(cur ^ 1, (kt + 1) * BK);      // prefetch next tile
      wait_async_le4();                        // retire current tile's 4 loads
    } else {
      wait_async_le0();
    }
    __syncthreads();                           // tile visible to all waves

    // ---- fragments per documented 16-bit A (16x32) / B (32x16) layouts ----
    Frag a[2], b[4];
#pragma unroll
    for (int tm = 0; tm < 2; ++tm) {
      int row = wm * 32 + tm * 16 + lrow;      // M row within block tile
      int kb  = lhi * 8;                       // K base for this lane half
      a[tm].u[0] = *(const uint4*)&As[cur][row][kb];      // K = kb..kb+7
      a[tm].u[1] = *(const uint4*)&As[cur][row][kb + 16]; // K = kb+16..kb+23
    }
#pragma unroll
    for (int tn = 0; tn < 4; ++tn) {
      int col = wn * 64 + tn * 16 + lrow;      // N column within block tile
      int ks  = lhi * 8;                       // K = ks..ks+7, ks+16..ks+23
      b[tn].u[0] = *(const uint4*)&Bs[cur][col][ks];
      b[tn].u[1] = *(const uint4*)&Bs[cur][col][ks + 16];
    }

    // ---- 8 WMMAs per wave per K step ----
#pragma unroll
    for (int tm = 0; tm < 2; ++tm)
#pragma unroll
      for (int tn = 0; tn < 4; ++tn)
        acc[tm][tn] = __builtin_amdgcn_wmma_f32_16x16x32_bf16(
            false, a[tm].v, false, b[tn].v, (short)0, acc[tm][tn],
            false, false);

    __syncthreads();                           // done reading `cur` buffer
  }

  // ---- store: D layout VGPR g -> M = g + 8*(lane>=16), N = lane&15 ----
#pragma unroll
  for (int tm = 0; tm < 2; ++tm) {
#pragma unroll
    for (int tn = 0; tn < 4; ++tn) {
      int col = tileN + wn * 64 + tn * 16 + lrow;
      if (col >= N) continue;
      float bv = bias ? bias[col] : 0.0f;
#pragma unroll
      for (int g = 0; g < 8; ++g) {
        int row = tileM + wm * 32 + tm * 16 + g + lhi * 8;
        float v = acc[tm][tn][g] + bv;
        size_t idx = (size_t)row * N + col;
        if (Cf)  Cf[idx]  = v;
        if (Cbf) Cbf[idx] = (__bf16)v;
      }
    }
  }
}

// ------------------------------------------------------- tiny attention
// One block per (b, h): S=23 queries/keys, HD=96. f32 in LDS, masked softmax.
__launch_bounds__(128)
__global__ void k_attention(const __bf16* __restrict__ qkv,
                            const int*    __restrict__ mask,
                            __bf16* __restrict__ ctx) {
  __shared__ float qs[S_][HD_];
  __shared__ float kk[S_][HD_];
  __shared__ float vv[S_][HD_];
  __shared__ float sc[S_][S_ + 1];
  __shared__ int   mk[S_];

  const int b   = blockIdx.x >> 3;
  const int h   = blockIdx.x & 7;
  const int tid = threadIdx.x;
  const size_t rowbase = (size_t)b * S_;
  const int qoff = h * HD_;
  const int koff = D_ + h * HD_;
  const int voff = 2 * D_ + h * HD_;

  if (tid < S_) mk[tid] = mask[b * S_ + tid];
  for (int i = tid; i < S_ * HD_; i += 128) {
    int s = i / HD_, d = i % HD_;
    size_t r = (rowbase + s) * TD_;
    qs[s][d] = (float)qkv[r + qoff + d];
    kk[s][d] = (float)qkv[r + koff + d];
    vv[s][d] = (float)qkv[r + voff + d];
  }
  __syncthreads();

  const float scale = 0.10206207261596575f;   // 1/sqrt(96)
  for (int p = tid; p < S_ * S_; p += 128) {
    int i = p / S_, j = p % S_;
    float a = 0.f;
#pragma unroll
    for (int d = 0; d < HD_; ++d) a += qs[i][d] * kk[j][d];
    sc[i][j] = mk[j] ? a * scale : -1e9f;
  }
  __syncthreads();

  if (tid < S_) {
    float mx = -1e30f;
    for (int j = 0; j < S_; ++j) mx = fmaxf(mx, sc[tid][j]);
    float sum = 0.f;
    for (int j = 0; j < S_; ++j) {
      float e = __expf(sc[tid][j] - mx);
      sc[tid][j] = e;
      sum += e;
    }
    float inv = 1.0f / sum;
    for (int j = 0; j < S_; ++j) sc[tid][j] *= inv;
  }
  __syncthreads();

  for (int o = tid; o < S_ * HD_; o += 128) {
    int i = o / HD_, d = o % HD_;
    float a = 0.f;
    for (int j = 0; j < S_; ++j) a += sc[i][j] * vv[j][d];
    ctx[(rowbase + i) * D_ + qoff + d] = (__bf16)a;
  }
}

// ------------------------------------------------- merge + section mean
// out = update ? attn_out : sec ; doc = mean_s(out) -> bf16 for pred GEMM
__global__ void k_merge_mean(const float* __restrict__ attn_out,
                             const float* __restrict__ cls,
                             const float* __restrict__ miss,
                             const int*   __restrict__ mask,
                             __bf16* __restrict__ docbf) {
  int i = blockIdx.x * blockDim.x + threadIdx.x;
  if (i >= B_ * D_) return;
  int b = i / D_, d = i % D_;
  int m[S_], any = 0;
#pragma unroll
  for (int s = 0; s < S_; ++s) { m[s] = mask[b * S_ + s]; any |= m[s]; }
  float sum = 0.f;
#pragma unroll
  for (int s = 0; s < S_; ++s) {
    size_t idx = ((size_t)b * S_ + s) * D_ + d;
    float secv = m[s] ? cls[idx] : miss[s * D_ + d];
    float v = (!m[s] && any) ? attn_out[idx] : secv;
    sum += v;
  }
  docbf[i] = (__bf16)(sum * (1.0f / (float)S_));
}

// ---------------------------------------------------------------- launcher
extern "C" void kernel_launch(void* const* d_in, const int* in_sizes, int n_in,
                              void* d_out, int out_size, void* d_ws,
                              size_t ws_size, hipStream_t stream) {
  (void)in_sizes; (void)n_in; (void)out_size; (void)ws_size;
  const float* cls   = (const float*)d_in[0];
  const float* miss  = (const float*)d_in[1];
  const float* wqkv  = (const float*)d_in[2];
  const float* bqkv  = (const float*)d_in[3];
  const float* wout  = (const float*)d_in[4];
  const float* bout  = (const float*)d_in[5];
  const float* wpred = (const float*)d_in[6];
  const float* bpred = (const float*)d_in[7];
  const int*   mask  = (const int*)d_in[8];
  float* logits = (float*)d_out;

  // workspace carve-up (256B aligned)
  char* ws = (char*)d_ws;
  size_t off = 0;
  auto take = [&](size_t bytes) -> char* {
    char* p = ws + off;
    off += (bytes + 255) & ~(size_t)255;
    return p;
  };
  const size_t BS = (size_t)B_ * S_;               // 94208
  __bf16* secbf   = (__bf16*)take(BS * D_ * 2);
  __bf16* wqkvbf  = (__bf16*)take((size_t)TD_ * D_ * 2);
  __bf16* woutbf  = (__bf16*)take((size_t)D_ * D_ * 2);
  __bf16* wpredbf = (__bf16*)take((size_t)L_ * D_ * 2);
  __bf16* qkvbf   = (__bf16*)take(BS * TD_ * 2);
  __bf16* ctxbf   = (__bf16*)take(BS * D_ * 2);
  float*  outf    = (float*) take(BS * D_ * 4);
  __bf16* docbf   = (__bf16*)take((size_t)B_ * D_ * 2);

  // 0) weights f32 -> bf16
  k_f32_to_bf16<<<(TD_ * D_ + 255) / 256, 256, 0, stream>>>(wqkv, wqkvbf, TD_ * D_);
  k_f32_to_bf16<<<(D_ * D_ + 255) / 256, 256, 0, stream>>>(wout, woutbf, D_ * D_);
  k_f32_to_bf16<<<(L_ * D_ + 255) / 256, 256, 0, stream>>>(wpred, wpredbf, L_ * D_);

  // 1) sec = mask-select(cls, missing_table) -> bf16
  int totalSec = (int)(BS * D_);
  k_build_sec<<<(totalSec + 255) / 256, 256, 0, stream>>>(cls, miss, mask,
                                                          secbf, totalSec);

  // 2) qkv = sec @ in_proj_w^T + b    (94208 x 2304 x 768) -> bf16
  dim3 g1(TD_ / BN, (int)(BS / BM));
  k_gemm_bf16<<<g1, 256, 0, stream>>>(secbf, wqkvbf, bqkv, nullptr, qkvbf,
                                      (int)BS, TD_, D_);

  // 3) per-(b,h) masked attention -> ctx bf16
  k_attention<<<B_ * H_, 128, 0, stream>>>(qkvbf, mask, ctxbf);

  // 4) attn_out = ctx @ out_proj_w^T + b   (94208 x 768 x 768) -> f32
  dim3 g2(D_ / BN, (int)(BS / BM));
  k_gemm_bf16<<<g2, 256, 0, stream>>>(ctxbf, woutbf, bout, outf, nullptr,
                                      (int)BS, D_, D_);

  // 5) merge missing-section updates + mean over sections -> doc bf16
  k_merge_mean<<<(B_ * D_ + 255) / 256, 256, 0, stream>>>(outf, cls, miss,
                                                          mask, docbf);

  // 6) logits = doc @ pred_w^T + b   (4096 x 50 x 768), ragged N handled
  dim3 g3((L_ + BN - 1) / BN, B_ / BM);
  k_gemm_bf16<<<g3, 256, 0, stream>>>(docbf, wpredbf, bpred, logits, nullptr,
                                      B_, L_, D_);
}
